// ThermoGibbsSampler_82549271429683
// MI455X (gfx1250) — matrix operand order repaired and checked by reading, not compiled, not run on hardware
//
#include <hip/hip_runtime.h>
#include <hip/hip_bf16.h>

typedef __attribute__((ext_vector_type(16))) _Float16 v16h;
typedef __attribute__((ext_vector_type(8)))  _Float16 v8h;
typedef __attribute__((ext_vector_type(8)))  float    v8f;

union Frag16 { v16h v; v8h h[2]; };

// A fragment (16x32 f16): lane<16 holds K {0..7,16..23}, lane>=16 holds K {8..15,24..31}
static __device__ __forceinline__ v16h frag_a(const _Float16* __restrict__ p, int lane) {
    Frag16 f;
    const int ko = (lane & 16) ? 8 : 0;
    f.h[0] = *(const v8h*)(p + ko);
    f.h[1] = *(const v8h*)(p + ko + 16);
    return f.v;
}
// B fragment (32x16 f16): lane<16 holds K {0..7,16..23} of column (lane&15), lane>=16 holds K {8..15,24..31}
static __device__ __forceinline__ v16h frag_b(const _Float16* __restrict__ p, int lane) {
    Frag16 f;
    const int ko = (lane & 16) ? 8 : 0;
    f.h[0] = *(const v8h*)(p + ko);
    f.h[1] = *(const v8h*)(p + ko + 16);
    return f.v;
}

// ---------------------------------------------------------------------------
// f32 -> f16 convert
__global__ void f32_to_f16(const float* __restrict__ in, _Float16* __restrict__ out, long long n) {
    long long i = (long long)blockIdx.x * blockDim.x + threadIdx.x;
    long long stride = (long long)gridDim.x * blockDim.x;
    for (; i < n; i += stride) out[i] = (_Float16)in[i];
}

// Vt[b][d][k] = V[b][k][d], f32 -> f16
__global__ void transpose_v(const float* __restrict__ vin, _Float16* __restrict__ vt,
                            int Bn, int KVn, int Dn) {
    long long n = (long long)Bn * KVn * Dn;
    for (long long i = (long long)blockIdx.x * blockDim.x + threadIdx.x; i < n;
         i += (long long)gridDim.x * blockDim.x) {
        int  kk = (int)(i % KVn);
        long long t = i / KVn;
        int  d  = (int)(t % Dn);
        int  b  = (int)(t / Dn);
        vt[i] = (_Float16)vin[((long long)b * KVn + kk) * Dn + d];
    }
}

// ---------------------------------------------------------------------------
// Generic NT GEMM: C[b,m,n] = epi( scale * sum_k A[b,m,k]*Bm[b,n,k] )
// epi: 0 = f16 raw, 1 = f16 scaled, 2 = f16 tanh(scale*x), 3 = f32 raw
__global__ void gemm_nt_f16(const _Float16* __restrict__ A,
                            const _Float16* __restrict__ Bm,
                            void* __restrict__ Cout,
                            int M, int N, int Kd,
                            long long sA, long long sB, long long sC,
                            float scale, int epi) {
    const int b    = blockIdx.y;
    A  += (size_t)b * sA;
    Bm += (size_t)b * sB;
    const int lane = threadIdx.x & 31;
    const int wave = threadIdx.x >> 5;
    const int ntn    = N >> 4;
    const int ntiles = (M >> 4) * ntn;
    const int tile   = blockIdx.x * (blockDim.x >> 5) + wave;
    if (tile >= ntiles) return;               // wave-uniform exit (EXEC all-1 for WMMA)
    const int tm = tile / ntn, tn = tile % ntn;

    const _Float16* arow = A  + (size_t)(tm * 16 + (lane & 15)) * Kd;
    const _Float16* brow = Bm + (size_t)(tn * 16 + (lane & 15)) * Kd;

    v8f acc = {};
    for (int kk = 0; kk < Kd; kk += 32) {
        v16h av = frag_a(arow + kk, lane);
        v16h bv = frag_b(brow + kk, lane);
        acc = __builtin_amdgcn_wmma_f32_16x16x32_f16(false, av, false, bv,
                                                     (short)0, acc, false, false);
    }

    const int col   = tn * 16 + (lane & 15);
    const int rbase = tm * 16 + ((lane & 16) ? 8 : 0);
    if (epi == 3) {
        float* C = (float*)Cout + (size_t)b * sC;
#pragma unroll
        for (int r = 0; r < 8; ++r)
            C[(size_t)(rbase + r) * N + col] = acc[r];
    } else {
        _Float16* C = (_Float16*)Cout + (size_t)b * sC;
#pragma unroll
        for (int r = 0; r < 8; ++r) {
            float v = acc[r];
            if (epi >= 1) v *= scale;
            if (epi == 2) v = tanhf(v);
            C[(size_t)(rbase + r) * N + col] = (_Float16)v;
        }
    }
}

// ---------------------------------------------------------------------------
// Row softmax: m[row,:] = softmax(h[row,:]) over N columns (f16 in / f16 out)
__global__ void softmax_rows(const _Float16* __restrict__ h, _Float16* __restrict__ m, int N) {
    const int row = blockIdx.x;
    const _Float16* hr = h + (size_t)row * N;
    _Float16*       mr = m + (size_t)row * N;
    __shared__ float red[256];
    float mx = -3.0e38f;
    for (int i = threadIdx.x; i < N; i += blockDim.x) mx = fmaxf(mx, (float)hr[i]);
    red[threadIdx.x] = mx; __syncthreads();
    for (int s = 128; s > 0; s >>= 1) {
        if ((int)threadIdx.x < s) red[threadIdx.x] = fmaxf(red[threadIdx.x], red[threadIdx.x + s]);
        __syncthreads();
    }
    mx = red[0]; __syncthreads();
    float sum = 0.f;
    for (int i = threadIdx.x; i < N; i += blockDim.x) sum += __expf((float)hr[i] - mx);
    red[threadIdx.x] = sum; __syncthreads();
    for (int s = 128; s > 0; s >>= 1) {
        if ((int)threadIdx.x < s) red[threadIdx.x] += red[threadIdx.x + s];
        __syncthreads();
    }
    const float rinv = 1.0f / red[0];
    for (int i = threadIdx.x; i < N; i += blockDim.x)
        mr[i] = (_Float16)(__expf((float)hr[i] - mx) * rinv);
}

// ---------------------------------------------------------------------------
// Fused TAP step: m_next = softmax_row( beta * (m_prev @ J + h) )
// Jt is J transposed (Jt[j,k] = J[k,j]) so the GEMM is NT.
// 16 waves; workgroup computes a 32(row) x 2048(col) strip; each wave owns a
// 32x128 patch (2 M-tiles x 8 N-tiles -> 16 WMMA accumulators); B fragments are
// reused by two WMMAs to raise compute/load ratio; softmax fused in-kernel.
__global__ __launch_bounds__(512) void tap_step(const _Float16* __restrict__ mprev,
                                                const _Float16* __restrict__ Jt,
                                                const _Float16* __restrict__ hmat,
                                                _Float16* __restrict__ mnext,
                                                const float* __restrict__ lbo,
                                                float beta_base, int Qn, int KVn) {
    const int b    = blockIdx.y;
    const int q0   = blockIdx.x * 32;
    const int lane = threadIdx.x & 31;
    const int wave = threadIdx.x >> 5;                 // 0..15, owns cols [wave*128, wave*128+128)
    const size_t qk  = (size_t)Qn * KVn;
    const size_t kk2 = (size_t)KVn * KVn;
    const _Float16* mp = mprev + (size_t)b * qk;
    const _Float16* hp = hmat  + (size_t)b * qk;
    const _Float16* jp = Jt    + (size_t)b * kk2;
    _Float16*       mo = mnext + (size_t)b * qk;

    // beta_t = base + sigmoid(log_beta_offset)*(beta_end-beta_start)
    const float beta = beta_base + 0.4f / (1.0f + __expf(-lbo[0]));

    v8f acc[2][8] = {};
    const _Float16* arow0 = mp + (size_t)(q0 + (lane & 15)) * KVn;
    const _Float16* arow1 = arow0 + (size_t)16 * KVn;
    const int n_lane0 = wave * 128 + (lane & 15);
    const _Float16* jrow0 = jp + (size_t)n_lane0 * KVn;

    for (int kk = 0; kk < KVn; kk += 32) {
        __builtin_prefetch(arow0 + kk + 32, 0, 0);
        __builtin_prefetch(jrow0 + kk + 32, 0, 0);
        v16h av0 = frag_a(arow0 + kk, lane);
        v16h av1 = frag_a(arow1 + kk, lane);
#pragma unroll
        for (int t = 0; t < 8; ++t) {
            v16h bv = frag_b(jrow0 + (size_t)(t * 16) * KVn + kk, lane);
            acc[0][t] = __builtin_amdgcn_wmma_f32_16x16x32_f16(false, av0, false, bv,
                                                               (short)0, acc[0][t], false, false);
            acc[1][t] = __builtin_amdgcn_wmma_f32_16x16x32_f16(false, av1, false, bv,
                                                               (short)0, acc[1][t], false, false);
        }
    }

    const int rbase   = (lane & 16) ? 8 : 0;           // C layout: VGPR r -> row (mt*16 + rbase + r)
    const int colbase = wave * 128 + (lane & 15);

    // x = beta * (mJ + h)
#pragma unroll
    for (int mt = 0; mt < 2; ++mt)
#pragma unroll
        for (int t = 0; t < 8; ++t)
#pragma unroll
            for (int r = 0; r < 8; ++r) {
                float hv = (float)hp[(size_t)(q0 + mt * 16 + rbase + r) * KVn + colbase + t * 16];
                acc[mt][t][r] = beta * (acc[mt][t][r] + hv);
            }

    __shared__ float pred[16][32];   // [wave][row]
    __shared__ float rowred[32];

    // --- row max: register reduce over 8 tiles, butterfly over the 16-lane half ---
    float vmax[2][8];
#pragma unroll
    for (int mt = 0; mt < 2; ++mt)
#pragma unroll
        for (int r = 0; r < 8; ++r) {
            float mx = acc[mt][0][r];
#pragma unroll
            for (int t = 1; t < 8; ++t) mx = fmaxf(mx, acc[mt][t][r]);
            vmax[mt][r] = mx;
        }
#pragma unroll
    for (int off = 8; off >= 1; off >>= 1)
#pragma unroll
        for (int mt = 0; mt < 2; ++mt)
#pragma unroll
            for (int r = 0; r < 8; ++r)
                vmax[mt][r] = fmaxf(vmax[mt][r], __shfl_xor(vmax[mt][r], off, 32));
    if ((lane & 15) == 0) {
#pragma unroll
        for (int mt = 0; mt < 2; ++mt)
#pragma unroll
            for (int r = 0; r < 8; ++r) pred[wave][mt * 16 + rbase + r] = vmax[mt][r];
    }
    __syncthreads();
    if (threadIdx.x < 32) {
        float mx = pred[0][threadIdx.x];
        for (int w2 = 1; w2 < 16; ++w2) mx = fmaxf(mx, pred[w2][threadIdx.x]);
        rowred[threadIdx.x] = mx;
    }
    __syncthreads();
    float rmax[2][8];
#pragma unroll
    for (int mt = 0; mt < 2; ++mt)
#pragma unroll
        for (int r = 0; r < 8; ++r) rmax[mt][r] = rowred[mt * 16 + rbase + r];

    // --- exp + row sum ---
    float vsum[2][8];
#pragma unroll
    for (int mt = 0; mt < 2; ++mt)
#pragma unroll
        for (int r = 0; r < 8; ++r) vsum[mt][r] = 0.f;
#pragma unroll
    for (int mt = 0; mt < 2; ++mt)
#pragma unroll
        for (int t = 0; t < 8; ++t)
#pragma unroll
            for (int r = 0; r < 8; ++r) {
                float e = __expf(acc[mt][t][r] - rmax[mt][r]);
                acc[mt][t][r] = e;
                vsum[mt][r] += e;
            }
#pragma unroll
    for (int off = 8; off >= 1; off >>= 1)
#pragma unroll
        for (int mt = 0; mt < 2; ++mt)
#pragma unroll
            for (int r = 0; r < 8; ++r)
                vsum[mt][r] += __shfl_xor(vsum[mt][r], off, 32);
    if ((lane & 15) == 0) {
#pragma unroll
        for (int mt = 0; mt < 2; ++mt)
#pragma unroll
            for (int r = 0; r < 8; ++r) pred[wave][mt * 16 + rbase + r] = vsum[mt][r];
    }
    __syncthreads();
    if (threadIdx.x < 32) {
        float s = pred[0][threadIdx.x];
        for (int w2 = 1; w2 < 16; ++w2) s += pred[w2][threadIdx.x];
        rowred[threadIdx.x] = 1.0f / s;
    }
    __syncthreads();
    float rinv[2][8];
#pragma unroll
    for (int mt = 0; mt < 2; ++mt)
#pragma unroll
        for (int r = 0; r < 8; ++r) rinv[mt][r] = rowred[mt * 16 + rbase + r];

    // --- write m_next (f16) ---
#pragma unroll
    for (int mt = 0; mt < 2; ++mt)
#pragma unroll
        for (int t = 0; t < 8; ++t)
#pragma unroll
            for (int r = 0; r < 8; ++r)
                mo[(size_t)(q0 + mt * 16 + rbase + r) * KVn + colbase + t * 16] =
                    (_Float16)(acc[mt][t][r] * rinv[mt][r]);
}

// ---------------------------------------------------------------------------
extern "C" void kernel_launch(void* const* d_in, const int* in_sizes, int n_in,
                              void* d_out, int out_size, void* d_ws, size_t ws_size,
                              hipStream_t stream) {
    const int B = 8, Q = 2048, KV = 2048, D = 128;
    const float scale = 0.08838834764831845f;  // 1/sqrt(128)

    const float* q   = (const float*)d_in[0];
    const float* k   = (const float*)d_in[1];
    const float* v   = (const float*)d_in[2];
    const float* wj  = (const float*)d_in[3];
    const float* lbo = (const float*)d_in[4];
    float* out = (float*)d_out;

    char* ws = (char*)d_ws;
    size_t off = 0;
    auto take = [&](size_t bytes) -> char* {
        char* p = ws + off;
        off += (bytes + 255) & ~(size_t)255;
        return p;
    };
    _Float16* Qh  = (_Float16*)take((size_t)B * Q  * D  * 2);
    _Float16* Kh  = (_Float16*)take((size_t)B * KV * D  * 2);
    _Float16* Wh  = (_Float16*)take((size_t)D * D * 2);
    _Float16* Jkh = (_Float16*)take((size_t)B * KV * D  * 2);
    _Float16* Vt  = (_Float16*)take((size_t)B * D  * KV * 2);
    _Float16* Hm  = (_Float16*)take((size_t)B * Q  * KV * 2);
    _Float16* Jt  = (_Float16*)take((size_t)B * KV * KV * 2);
    _Float16* M0  = (_Float16*)take((size_t)B * Q  * KV * 2);
    _Float16* M1  = (_Float16*)take((size_t)B * Q  * KV * 2);

    auto cvt = [&](const float* src, _Float16* dst, long long n) {
        f32_to_f16<<<(int)((n + 255) / 256), 256, 0, stream>>>(src, dst, n);
    };
    cvt(q,  Qh, (long long)B * Q  * D);
    cvt(k,  Kh, (long long)B * KV * D);
    cvt(wj, Wh, (long long)D * D);
    {
        long long n = (long long)B * D * KV;
        transpose_v<<<(int)((n + 255) / 256), 256, 0, stream>>>(v, Vt, B, KV, D);
    }

    auto gemm = [&](const _Float16* A, const _Float16* Bm, void* C,
                    int M, int N, int Kd, long long sA, long long sB, long long sC,
                    float sc, int epi) {
        int ntiles = (M / 16) * (N / 16);
        dim3 g((ntiles + 7) / 8, B);
        gemm_nt_f16<<<g, 256, 0, stream>>>(A, Bm, C, M, N, Kd, sA, sB, sC, sc, epi);
    };

    // Jk[b,j,d] = sum_e K[b,j,e] * W_J[d,e]
    gemm(Kh, Wh, Jkh, KV, D, D, (long long)KV * D, 0, (long long)KV * D, 1.0f, 0);
    // H[b,q,j] = scale * Q_q . K_j
    gemm(Qh, Kh, Hm, Q, KV, D, (long long)Q * D, (long long)KV * D, (long long)Q * KV, scale, 1);
    // Jt[b,j,k] = tanh(scale * Jk_j . K_k)   ( == J[k,j] )
    gemm(Jkh, Kh, Jt, KV, KV, D, (long long)KV * D, (long long)KV * D, (long long)KV * KV, scale, 2);
    // m0 = softmax(H)
    softmax_rows<<<B * Q, 256, 0, stream>>>(Hm, M0, KV);

    _Float16* mcur = M0;
    _Float16* mnxt = M1;
    for (int step = 0; step < 4; ++step) {
        float beta_base = 0.8f + (step / 3.0f) * 0.4f;
        dim3 g(Q / 32, B);
        tap_step<<<g, 512, 0, stream>>>(mcur, Jt, Hm, mnxt, lbo, beta_base, Q, KV);
        _Float16* t = mcur; mcur = mnxt; mnxt = t;
    }

    // out[b,q,d] = sum_k m[b,q,k] * V[b,k,d]  (f32 output)
    gemm(mcur, Vt, out, Q, D, KV, (long long)Q * KV, (long long)D * KV, (long long)Q * D, 1.0f, 3);

    (void)in_sizes; (void)n_in; (void)out_size; (void)ws_size;
}
